// PleLayer_16819091931553
// MI455X (gfx1250) — matrix operand order repaired and checked by reading, not compiled, not run on hardware
//
#include <hip/hip_runtime.h>

typedef __attribute__((ext_vector_type(16))) __bf16 v16bf;
typedef __attribute__((ext_vector_type(8)))  __bf16 v8bf;
typedef __attribute__((ext_vector_type(8)))  float  v8f;

constexpr int B = 16384, D = 1024, E = 512, T = 2, NE = 4, NS = 4;
constexpr int KS = D / 32;                            // 32 k-steps of 32
constexpr int NT = E / 16;                            // 32 n-tiles per expert
constexpr int NEXP = NS + T * NE;                     // 12 experts
constexpr int FRAGS_PER_EXP = KS * NT;                // 1024
constexpr int GATE_FRAG_BASE = NEXP * FRAGS_PER_EXP;  // 12288
constexpr int TOTAL_FRAGS = GATE_FRAG_BASE + KS;      // 12320
constexpr int FRAG_ELEMS = 512;                       // one 16x32 bf16 fragment
constexpr long W_ELEMS = (long)TOTAL_FRAGS * FRAG_ELEMS;
// workspace requirement: 2 * W_ELEMS * 2 bytes = ~25.3 MB (fits easily in 192MB L2)

static __device__ __forceinline__ v8f wmma_bf16(v16bf a, v16bf b, v8f c) {
  return __builtin_amdgcn_wmma_f32_16x16x32_bf16(false, a, false, b, (short)0, c,
                                                 false, false);
}

// ---------------------------------------------------------------------------
// Prep: weights (f32) -> bf16 hi/lo, pre-swizzled into WMMA B-fragment layout.
// B 32x16 (KxN) as packed here: lane holds column n=lane&15; element j (0..15) -> k = 16*(lane>=16)+j.
// One thread per (frag, lane, half): 8 contiguous-k elements = one 16B store.
// ---------------------------------------------------------------------------
__global__ __launch_bounds__(256) void ple_prep(
    const float* __restrict__ Ws, const float* __restrict__ Wt,
    const float* __restrict__ Wg, __bf16* __restrict__ whi,
    __bf16* __restrict__ wlo) {
  long tid = (long)blockIdx.x * 256 + threadIdx.x;
  if (tid >= (long)TOTAL_FRAGS * 64) return;
  int half = (int)(tid & 1);
  int lane = (int)((tid >> 1) & 31);
  int frag = (int)(tid >> 6);
  int nloc = lane & 15;
  int k0 = ((lane >> 4) << 4) + half * 8;  // B-fragment K run start
  const float* src;
  long dstride;
  int ks;
  if (frag >= GATE_FRAG_BASE) {  // gate "expert": 16 cols = [task0 g0..7 | task1 g0..7]
    ks = frag - GATE_FRAG_BASE;
    int t_ = nloc >> 3, j = nloc & 7;
    src = Wg + (long)t_ * D * (NS + NE) + j;
    dstride = NS + NE;
  } else {
    int e = frag / FRAGS_PER_EXP;
    int rem = frag - e * FRAGS_PER_EXP;
    ks = rem >> 5;
    int nt = rem & 31;
    int ncol = nt * 16 + nloc;
    if (e < NS) src = Ws + (long)e * D * E + ncol;         // shared experts
    else        src = Wt + (long)(e - NS) * D * E + ncol;  // task experts (t*NE+j = e-NS)
    dstride = E;
  }
  int dbase = ks * 32 + k0;
  v8bf hv, lv;
#pragma unroll
  for (int q = 0; q < 8; ++q) {
    float w = src[(long)(dbase + q) * dstride];
    __bf16 h = (__bf16)w;
    hv[q] = h;
    lv[q] = (__bf16)(w - (float)h);
  }
  long off = (long)frag * FRAG_ELEMS + lane * 16 + half * 8;
  *(v8bf*)(whi + off) = hv;
  *(v8bf*)(wlo + off) = lv;
}

// ---------------------------------------------------------------------------
// Main fused kernel: 64 rows per workgroup, 8 waves.
// Wave w owns n-tiles w*4..w*4+3 (64 cols) x all 4 m-tiles (64 rows).
// ---------------------------------------------------------------------------
__global__ __launch_bounds__(256) void ple_main(
    const float* __restrict__ x, const float* __restrict__ bs,
    const float* __restrict__ bt, const float* __restrict__ bg,
    const __bf16* __restrict__ whi, const __bf16* __restrict__ wlo,
    float* __restrict__ out) {
  // A fragments (16x32 bf16 per (m-tile,k-step)), fragment-major for conflict-free loads
  __shared__ __align__(16) __bf16 sAhi[4 * 32 * 32 * 16];  // 128 KB
  __shared__ __align__(16) __bf16 sAlo[4 * 32 * 32 * 16];  // 128 KB
  __shared__ float sG[64 * 16];                            // gate logits -> weights

  const int tid = threadIdx.x;
  const int lane = tid & 31;
  const int wave = tid >> 5;
  const long row0 = (long)blockIdx.x * 64;

  // ---- phase 0: load x block once, split to bf16 hi/lo A-fragments in LDS ----
  // A (16x32): lane = m (+16 for second k-half); run of 8 consecutive k per (lane,half).
  for (int it = 0; it < 32; ++it) {
    int idx = it * 256 + tid;
    int run = idx & 127;  // 8-element d-run within the row
    int row = idx >> 7;   // 0..63
    int ks = run >> 2;
    int sub = run & 3;
    int flane = (row & 15) + ((sub >> 1) << 4);
    int jhalf = sub & 1;
    int mt = row >> 4;
    const float* xp = x + (row0 + row) * D + run * 8;
    long foff = ((long)((mt * 32 + ks) * 32 + flane)) * 16 + jhalf * 8;
    v8bf hv, lv;
#pragma unroll
    for (int q = 0; q < 8; ++q) {
      float f = xp[q];
      __bf16 h = (__bf16)f;
      hv[q] = h;
      lv[q] = (__bf16)(f - (float)h);
    }
    *(v8bf*)(sAhi + foff) = hv;
    *(v8bf*)(sAlo + foff) = lv;
  }
  __syncthreads();

  // ---- phase 1: gate logits via WMMA (waves 0..3, one m-tile each) ----
  if (wave < 4) {
    int mt = wave;
    v8f acc = {};
    const __bf16* gbh = whi + (long)GATE_FRAG_BASE * FRAG_ELEMS + lane * 16;
    const __bf16* gbl = wlo + (long)GATE_FRAG_BASE * FRAG_ELEMS + lane * 16;
    for (int ks = 0; ks < KS; ++ks) {
      long aoff = ((long)((mt * 32 + ks) * 32 + lane)) * 16;
      v16bf ah = *(const v16bf*)(sAhi + aoff);
      v16bf al = *(const v16bf*)(sAlo + aoff);
      v16bf bh = *(const v16bf*)(gbh + (long)ks * FRAG_ELEMS);
      v16bf bl = *(const v16bf*)(gbl + (long)ks * FRAG_ELEMS);
      acc = wmma_bf16(ah, bh, acc);
      acc = wmma_bf16(ah, bl, acc);
      acc = wmma_bf16(al, bh, acc);
    }
    int col = lane & 15;         // gate index = t*8+k
    float bgv = bg[col];         // bg is (T, 8) flat = 16
    int rbase = mt * 16 + ((lane >> 4) << 3);
#pragma unroll
    for (int r = 0; r < 8; ++r) sG[(rbase + r) * 16 + col] = acc[r] + bgv;
  }
  __syncthreads();

  // ---- phase 2: softmax over 8 gates per (row, task), in place ----
  if (tid < 128) {
    int row = tid >> 1, t_ = tid & 1;
    float* gp = sG + row * 16 + t_ * 8;
    float m = gp[0];
#pragma unroll
    for (int k = 1; k < 8; ++k) m = fmaxf(m, gp[k]);
    float ev[8], s = 0.f;
#pragma unroll
    for (int k = 0; k < 8; ++k) { ev[k] = __expf(gp[k] - m); s += ev[k]; }
    float inv = 1.f / s;
#pragma unroll
    for (int k = 0; k < 8; ++k) gp[k] = ev[k] * inv;
  }
  __syncthreads();

  // ---- phase 3: 12 experts, bf16x3 WMMA GEMM + fused bias/relu/gated-combine ----
  const v8f vzero = {};
  v8f tower[2][4][4];
#pragma unroll
  for (int t_ = 0; t_ < 2; ++t_)
#pragma unroll
    for (int mt = 0; mt < 4; ++mt)
#pragma unroll
      for (int i = 0; i < 4; ++i) tower[t_][mt][i] = vzero;

  const int col_l = lane & 15;
  const int rsel = (lane >> 4) << 3;

  for (int e = 0; e < NEXP; ++e) {
    v8f eacc[4][4];
#pragma unroll
    for (int mt = 0; mt < 4; ++mt)
#pragma unroll
      for (int i = 0; i < 4; ++i) eacc[mt][i] = vzero;

    const __bf16* bhb = whi + (long)e * FRAGS_PER_EXP * FRAG_ELEMS + lane * 16;
    const __bf16* blb = wlo + (long)e * FRAGS_PER_EXP * FRAG_ELEMS + lane * 16;

    for (int ks = 0; ks < KS; ++ks) {
      v16bf ah[4], al[4];
#pragma unroll
      for (int mt = 0; mt < 4; ++mt) {
        long aoff = ((long)((mt * 32 + ks) * 32 + lane)) * 16;
        ah[mt] = *(const v16bf*)(sAhi + aoff);
        al[mt] = *(const v16bf*)(sAlo + aoff);
      }
#pragma unroll
      for (int i = 0; i < 4; ++i) {
        int nt = wave * 4 + i;
        long boff = (long)(ks * 32 + nt) * FRAG_ELEMS;
        v16bf bh = *(const v16bf*)(bhb + boff);
        v16bf bl = *(const v16bf*)(blb + boff);
#pragma unroll
        for (int mt = 0; mt < 4; ++mt) {
          eacc[mt][i] = wmma_bf16(ah[mt], bh, eacc[mt][i]);
          eacc[mt][i] = wmma_bf16(ah[mt], bl, eacc[mt][i]);
          eacc[mt][i] = wmma_bf16(al[mt], bh, eacc[mt][i]);
        }
      }
    }

    // epilogue: bias + relu + gate-weighted accumulation into towers
    float biasv[4];
#pragma unroll
    for (int i = 0; i < 4; ++i) {
      int col = (wave * 4 + i) * 16 + col_l;
      biasv[i] = (e < NS) ? bs[e * E + col] : bt[(long)(e - NS) * E + col];
    }
    // gate vector per (row,t) is [shared g0..3 | task g0..3] == sG[row*16 + t*8 + k]
    int g0i = e;                             // task-0 gate index (valid when e<8)
    int g1i = (e < NS) ? (8 + e) : (e + 4);  // task-1 gate index (valid when e<4 or e>=8)
    bool v0 = (e < 8);
    bool v1 = (e < NS) || (e >= 8);
#pragma unroll
    for (int mt = 0; mt < 4; ++mt) {
#pragma unroll
      for (int r = 0; r < 8; ++r) {
        int row = mt * 16 + r + rsel;
        float g0 = v0 ? sG[row * 16 + g0i] : 0.f;
        float g1 = v1 ? sG[row * 16 + g1i] : 0.f;
#pragma unroll
        for (int i = 0; i < 4; ++i) {
          float val = fmaxf(eacc[mt][i][r] + biasv[i], 0.f);
          tower[0][mt][i][r] += g0 * val;
          tower[1][mt][i][r] += g1 * val;
        }
      }
    }
  }

  // ---- phase 4: store towers, output (T, B, E) ----
#pragma unroll
  for (int t_ = 0; t_ < 2; ++t_)
#pragma unroll
    for (int mt = 0; mt < 4; ++mt)
#pragma unroll
      for (int i = 0; i < 4; ++i) {
        int col = (wave * 4 + i) * 16 + col_l;
#pragma unroll
        for (int r = 0; r < 8; ++r) {
          int row = mt * 16 + r + rsel;
          out[((long)t_ * B + row0 + row) * E + col] = tower[t_][mt][i][r];
        }
      }
}

extern "C" void kernel_launch(void* const* d_in, const int* in_sizes, int n_in,
                              void* d_out, int out_size, void* d_ws, size_t ws_size,
                              hipStream_t stream) {
  const float* x  = (const float*)d_in[0];
  const float* Ws = (const float*)d_in[1];
  const float* bs = (const float*)d_in[2];
  const float* Wt = (const float*)d_in[3];
  const float* bt = (const float*)d_in[4];
  const float* Wg = (const float*)d_in[5];
  const float* bg = (const float*)d_in[6];
  float* out = (float*)d_out;

  __bf16* whi = (__bf16*)d_ws;
  __bf16* wlo = whi + W_ELEMS;  // total ws use: 2*W_ELEMS*2 B ~= 25.3 MB

  int prep_blocks = (int)(((long)TOTAL_FRAGS * 64) / 256);  // 3080
  ple_prep<<<prep_blocks, 256, 0, stream>>>(Ws, Wt, Wg, whi, wlo);
  ple_main<<<B / 64, 256, 0, stream>>>(x, bs, bt, bg, whi, wlo, out);
}
